// LeafBlockAttention_38749194944612
// MI455X (gfx1250) — compile-verified, matrix operand order checked by reading, not run on hardware
//
#include <hip/hip_runtime.h>
#include <hip/hip_bf16.h>
#include <math.h>

// ---------------- problem constants ----------------
#define NBLK   2048      // NB
#define LTOK   32        // L
#define LP1    33        // L+1
#define CDIM   256       // C (also K of both GEMMs)
#define NHEAD  8         // H
#define HDIM   32        // head dim
#define NTOK   65536     // N tokens
#define KVROWS (NBLK * LP1)   // 67584
#define QKVN   768            // 3*C
#define VPAD   72             // padded LDS row length (144 B, 16B-aligned, bank-skewed)

typedef __attribute__((ext_vector_type(16))) __bf16 v16bf;
typedef __attribute__((ext_vector_type(8)))  float  v8f;
typedef __attribute__((ext_vector_type(4)))  unsigned int u32x4;
typedef __attribute__((ext_vector_type(8)))  int          i32x8;
typedef __attribute__((ext_vector_type(4)))  int          i32x4;

union FragU { v16bf v; uint4 q[2]; };
union V32U  { uint4 q[4]; __bf16 e[32]; };

// 16x32 bf16 A/B fragment per ISA 7.12.2: lane holds row (lane&15);
// elems 0..7 = K kbase..kbase+7, elems 8..15 = K kbase+16..kbase+23, kbase=(lane>>4)*8.
__device__ __forceinline__ v16bf frag_load(const __bf16* p, int kbase) {
    FragU u;
    u.q[0] = *(const uint4*)(p + kbase);
    u.q[1] = *(const uint4*)(p + kbase + 16);
    return u.v;
}
__device__ __forceinline__ v16bf frag_zero() {
    FragU u;
    u.q[0] = make_uint4(0u, 0u, 0u, 0u);
    u.q[1] = make_uint4(0u, 0u, 0u, 0u);
    return u.v;
}
__device__ __forceinline__ v8f wmma_bf16(v16bf a, v16bf b, v8f c) {
    return __builtin_amdgcn_wmma_f32_16x16x32_bf16(false, a, false, b, (short)0, c, false, false);
}

__device__ __forceinline__ void store_val(__bf16* p, float v) { *p = (__bf16)v; }
__device__ __forceinline__ void store_val(float*  p, float v) { *p = v; }

// ---- TDM: stage a contiguous bf16 panel (nelems elements) from global into LDS ----
// D# per CDNA5 ISA: group0 = {count/flags, lds_addr, gaddr_lo,
// gaddr_hi|type=2}; group1 packs data_size=2B, tensor_dim0 = tile_dim0 = nelems,
// tensor_dim0_stride = nelems (1D tile).
__device__ __forceinline__ void tdm_stage_panel(const __bf16* gsrc, __bf16* lds_dst,
                                                unsigned int nelems) {
#if __has_builtin(__builtin_amdgcn_tensor_load_to_lds)
#warning "CDNA5 probe: using __builtin_amdgcn_tensor_load_to_lds (TDM path)"
    unsigned long long gaddr = (unsigned long long)(uintptr_t)gsrc;
    unsigned int lds = (unsigned int)(uintptr_t)lds_dst;   // low 32 bits = LDS offset
    u32x4 g0;
    g0[0] = 1u;                                             // count=1, user descriptor
    g0[1] = lds;                                            // lds_addr
    g0[2] = (unsigned int)(gaddr & 0xFFFFFFFFu);            // global_addr[31:0]
    g0[3] = (unsigned int)((gaddr >> 32) & 0x1FFFFFFu) | (2u << 30); // addr[56:32], type=2
    i32x8 g1;
    g1[0] = (int)(1u << 16);                                // data_size=1 -> 2 bytes
    g1[1] = (int)((nelems & 0xFFFFu) << 16);                // tensor_dim0[15:0] @ bits 63:48
    g1[2] = (int)((nelems >> 16) & 0xFFFFu);                // tensor_dim0[31:16]; tensor_dim1=0
    g1[3] = (int)((nelems & 0xFFFFu) << 16);                // tile_dim0 @ bits 127:112
    g1[4] = 0;                                              // tile_dim1/2 unused
    g1[5] = (int)nelems;                                    // tensor_dim0_stride lo32
    g1[6] = 0;
    g1[7] = 0;
    i32x4 gz = {0, 0, 0, 0};
#if __has_include(<hip/amd_detail/amd_gfx1250_TDM.h>)
#warning "CDNA5 probe: 6-arg TDM builtin form (amdgpu-toolchain headers present)"
    i32x8 gz8 = {0, 0, 0, 0, 0, 0, 0, 0};
    __builtin_amdgcn_tensor_load_to_lds(g0, g1, gz, gz, gz8, 0);
#else
    __builtin_amdgcn_tensor_load_to_lds(g0, g1, gz, gz, 0);
#endif
#if __has_builtin(__builtin_amdgcn_s_wait_tensorcnt)
    __builtin_amdgcn_s_wait_tensorcnt(0);
#endif
#else
#warning "CDNA5 probe: TDM builtin ABSENT, using cooperative-copy fallback"
    // fallback: cooperative copy by this wave
    for (unsigned int i = (threadIdx.x & 31); i < nelems / 8u; i += 32u)
        ((uint4*)lds_dst)[i] = ((const uint4*)gsrc)[i];
#endif
}

// ---------------- kernel 1: weights -> bf16, transposed (N-major) ----------------
__global__ __launch_bounds__(256) void prep_w_kernel(const float* __restrict__ Wqkv,
                                                     const float* __restrict__ Wproj,
                                                     __bf16* __restrict__ wqkvT,
                                                     __bf16* __restrict__ wprojT) {
    int bid = blockIdx.x;
    int t   = threadIdx.x;
    if (bid < QKVN) {
        wqkvT[(size_t)bid * CDIM + t] = (__bf16)Wqkv[(size_t)t * QKVN + bid];
    } else {
        int n = bid - QKVN;
        wprojT[(size_t)n * CDIM + t] = (__bf16)Wproj[(size_t)t * CDIM + n];
    }
}

// ---------------- kernel 2: build kv (32 rows + mean row), fp32 -> bf16 ----------------
__global__ __launch_bounds__(256) void build_kv_kernel(const float* __restrict__ x,
                                                       __bf16* __restrict__ kv) {
    int b = blockIdx.x;
    int c = threadIdx.x;
    float sum = 0.0f;
    #pragma unroll 4
    for (int r = 0; r < LTOK; ++r) {
        float v = x[((size_t)b * LTOK + r) * CDIM + c];
        kv[((size_t)b * LP1 + r) * CDIM + c] = (__bf16)v;
        sum += v;
    }
    kv[((size_t)b * LP1 + LTOK) * CDIM + c] = (__bf16)(sum * (1.0f / 32.0f));
}

// ---------------- block GEMM: out(M x N) = A(M x 256) @ BwT(N x 256)^T + bias ----------
// Workgroup = 8 waves -> 128x64 block tile. Weight panel (64 rows x 256 K, 32 KB,
// contiguous since BwT is N-major) is TDM-staged into LDS once. Each wave computes a
// 32x32 tile: 2x2 accumulators, 4 WMMAs per 32-wide K step, K unrolled x8 = 32 WMMAs.
template <typename OutT>
__global__ __launch_bounds__(256) void gemm_block_kernel(const __bf16* __restrict__ A,
                                                         const __bf16* __restrict__ BwT,
                                                         const float* __restrict__ bias,
                                                         OutT* __restrict__ out,
                                                         int ldout) {
    __shared__ __bf16 Bsh[64 * CDIM];   // 32 KB

    const int wave = threadIdx.x >> 5;
    const int lane = threadIdx.x & 31;
    const int lr   = lane & 15;
    const int half = lane >> 4;
    const int kbase = half * 8;

    const int mbase = blockIdx.x * 128;
    const int nbase = blockIdx.y * 64;

    if (threadIdx.x < 32)
        tdm_stage_panel(BwT + (size_t)nbase * CDIM, Bsh, 64u * CDIM);
    __syncthreads();

    const int mwave = wave >> 1;     // 0..3 -> 32-row strip
    const int nwave = wave & 1;      // 0..1 -> 32-col strip

    const __bf16* arow0 = A + (size_t)(mbase + mwave * 32 + lr)      * CDIM;
    const __bf16* arow1 = A + (size_t)(mbase + mwave * 32 + 16 + lr) * CDIM;
    const __bf16* brow0 = &Bsh[(size_t)(nwave * 32 + lr)      * CDIM];
    const __bf16* brow1 = &Bsh[(size_t)(nwave * 32 + 16 + lr) * CDIM];

    v8f acc00 = {0.f,0.f,0.f,0.f,0.f,0.f,0.f,0.f};
    v8f acc01 = acc00, acc10 = acc00, acc11 = acc00;

    #pragma unroll
    for (int kt = 0; kt < CDIM / 32; ++kt) {
        v16bf a0 = frag_load(arow0 + kt * 32, kbase);
        v16bf a1 = frag_load(arow1 + kt * 32, kbase);
        v16bf b0 = frag_load(brow0 + kt * 32, kbase);
        v16bf b1 = frag_load(brow1 + kt * 32, kbase);
        acc00 = wmma_bf16(a0, b0, acc00);
        acc01 = wmma_bf16(a0, b1, acc01);
        acc10 = wmma_bf16(a1, b0, acc10);
        acc11 = wmma_bf16(a1, b1, acc11);
    }

    const int ncol0 = nbase + nwave * 32 + lr;
    const int ncol1 = ncol0 + 16;
    const float bias0 = bias[ncol0];
    const float bias1 = bias[ncol1];

    #pragma unroll
    for (int r = 0; r < 8; ++r) {
        const int row0 = mbase + mwave * 32 + r + 8 * half;
        const int row1 = row0 + 16;
        store_val(out + (size_t)row0 * ldout + ncol0, acc00[r] + bias0);
        store_val(out + (size_t)row0 * ldout + ncol1, acc01[r] + bias1);
        store_val(out + (size_t)row1 * ldout + ncol0, acc10[r] + bias0);
        store_val(out + (size_t)row1 * ldout + ncol1, acc11[r] + bias1);
    }
}

// ---------------- fused attention per (block, head), one wave32 ----------------
__global__ __launch_bounds__(32) void attn_kernel(const __bf16* __restrict__ qkv,
                                                  const float* __restrict__ attn_mask,
                                                  const float* __restrict__ edge,
                                                  const float* __restrict__ Wgate,
                                                  const float* __restrict__ bgate,
                                                  __bf16* __restrict__ aout) {
    __shared__ float  s[LTOK][48];        // raw scores / softmax scratch
    __shared__ __bf16 comb[LTOK][VPAD];   // attn + gate, K padded 33->64 (rows padded to 72)
    __shared__ __bf16 vT[HDIM][VPAD];     // v transposed per head

    int b = blockIdx.x;
    int h = blockIdx.y;
    int lane = threadIdx.x;
    int lr   = lane & 15;
    int half = lane >> 4;
    int kbase = half * 8;
    const float scale = 0.17677669529663687f;   // 32^-0.5

    const __bf16* qkv_blk = qkv + (size_t)b * LP1 * QKVN;

    // ---- zero vT with vectorized LDS stores (32 rows * 72 elems = 288 uint4) ----
    {
        uint4 z4 = make_uint4(0u, 0u, 0u, 0u);
        uint4* vz = (uint4*)&vT[0][0];
        #pragma unroll
        for (int i = 0; i < 9; ++i)
            vz[lane * 9 + i] = z4;
    }

    // ---- stage v^T: lane loads its v row wide (4 x b128), transposes via LDS ----
    {
        int kk = lane;                                   // key row 0..31
        const __bf16* vrow = qkv_blk + (size_t)kk * QKVN + 2 * CDIM + h * HDIM;
        V32U u;
        u.q[0] = *(const uint4*)(vrow);
        u.q[1] = *(const uint4*)(vrow + 8);
        u.q[2] = *(const uint4*)(vrow + 16);
        u.q[3] = *(const uint4*)(vrow + 24);
        #pragma unroll
        for (int d = 0; d < HDIM; ++d)
            vT[d][kk] = u.e[d];
        if (lane == 0) {                                 // summary row 32
            const __bf16* vrow32 = qkv_blk + (size_t)LTOK * QKVN + 2 * CDIM + h * HDIM;
            V32U u2;
            u2.q[0] = *(const uint4*)(vrow32);
            u2.q[1] = *(const uint4*)(vrow32 + 8);
            u2.q[2] = *(const uint4*)(vrow32 + 16);
            u2.q[3] = *(const uint4*)(vrow32 + 24);
            #pragma unroll
            for (int d = 0; d < HDIM; ++d)
                vT[d][LTOK] = u2.e[d];
        }
    }

    // ---- scores = q @ k^T  (M=32, N=33 pad 48, K=32) : 6 WMMAs ----
    v16bf afr[2];
    #pragma unroll
    for (int mt = 0; mt < 2; ++mt)
        afr[mt] = frag_load(qkv_blk + (size_t)(mt * 16 + lr) * QKVN + h * HDIM, kbase);
    v16bf bfr[3];
    #pragma unroll
    for (int nt = 0; nt < 3; ++nt) {
        int col = nt * 16 + lr;
        bfr[nt] = (col < LP1)
            ? frag_load(qkv_blk + (size_t)col * QKVN + CDIM + h * HDIM, kbase)
            : frag_zero();
    }
    v8f sc[2][3];
    #pragma unroll
    for (int mt = 0; mt < 2; ++mt)
        #pragma unroll
        for (int nt = 0; nt < 3; ++nt) {
            v8f z = {0.f,0.f,0.f,0.f,0.f,0.f,0.f,0.f};
            sc[mt][nt] = wmma_bf16(afr[mt], bfr[nt], z);
        }
    #pragma unroll
    for (int mt = 0; mt < 2; ++mt)
        #pragma unroll
        for (int nt = 0; nt < 3; ++nt)
            #pragma unroll
            for (int r = 0; r < 8; ++r)
                s[mt * 16 + r + 8 * half][nt * 16 + lr] = sc[mt][nt][r];

    __syncthreads();

    // ---- softmax + physics bias + edge gate (lane = query row) ----
    {
        int r = lane;
        const float* mrow = attn_mask + ((size_t)b * LTOK + r) * LP1;
        const float* erow = edge + ((size_t)b * LTOK + r) * LP1 * 4;
        float mx = -3.0e38f;
        for (int k = 0; k < LP1; ++k) {
            float wk = erow[k * 4 + 3];
            if (k == LTOK || k == r) wk = 1.0f;
            float sv = (mrow[k] == 0.0f) ? -3.0e38f : (s[r][k] * scale + wk);
            s[r][k] = sv;
            mx = fmaxf(mx, sv);
        }
        float sum = 0.0f;
        for (int k = 0; k < LP1; ++k) {
            float e = __expf(s[r][k] - mx);
            s[r][k] = e;
            sum += e;
        }
        float inv = 1.0f / sum;
        float wg = Wgate[h], bg = bgate[h];
        for (int k = 0; k < LP1; ++k) {
            float wk = erow[k * 4 + 3];
            if (k == LTOK || k == r) wk = 1.0f;
            float lew = (mrow[k] == 0.0f) ? 0.0f : (wk * wg + bg);
            comb[r][k] = (__bf16)(s[r][k] * inv + lew);
        }
        for (int k = LP1; k < 64; ++k)
            comb[r][k] = (__bf16)0.0f;
    }
    __syncthreads();

    // ---- out = combined(32x64) @ v(64x32) : 8 WMMAs ----
    v8f oc[2][2];
    #pragma unroll
    for (int mt = 0; mt < 2; ++mt)
        #pragma unroll
        for (int nt = 0; nt < 2; ++nt) {
            v8f z = {0.f,0.f,0.f,0.f,0.f,0.f,0.f,0.f};
            oc[mt][nt] = z;
        }
    #pragma unroll
    for (int kt = 0; kt < 2; ++kt) {
        v16bf af[2], bf2[2];
        #pragma unroll
        for (int mt = 0; mt < 2; ++mt)
            af[mt] = frag_load(&comb[mt * 16 + lr][kt * 32], kbase);
        #pragma unroll
        for (int nt = 0; nt < 2; ++nt)
            bf2[nt] = frag_load(&vT[nt * 16 + lr][kt * 32], kbase);
        #pragma unroll
        for (int mt = 0; mt < 2; ++mt)
            #pragma unroll
            for (int nt = 0; nt < 2; ++nt)
                oc[mt][nt] = wmma_bf16(af[mt], bf2[nt], oc[mt][nt]);
    }
    #pragma unroll
    for (int mt = 0; mt < 2; ++mt)
        #pragma unroll
        for (int nt = 0; nt < 2; ++nt)
            #pragma unroll
            for (int r = 0; r < 8; ++r) {
                int row = mt * 16 + r + 8 * half;
                int col = nt * 16 + lr;
                aout[((size_t)b * LTOK + row) * CDIM + h * HDIM + col] = (__bf16)oc[mt][nt][r];
            }
}

// ---------------- launch ----------------
extern "C" void kernel_launch(void* const* d_in, const int* in_sizes, int n_in,
                              void* d_out, int out_size, void* d_ws, size_t ws_size,
                              hipStream_t stream) {
    const float* x     = (const float*)d_in[0];
    const float* amask = (const float*)d_in[1];
    const float* edge  = (const float*)d_in[2];
    const float* Wqkv  = (const float*)d_in[3];
    const float* bqkv  = (const float*)d_in[4];
    const float* Wproj = (const float*)d_in[5];
    const float* bproj = (const float*)d_in[6];
    const float* Wgate = (const float*)d_in[7];
    const float* bgate = (const float*)d_in[8];
    float* out = (float*)d_out;

    char* ws = (char*)d_ws;
    size_t off = 0;
    __bf16* kv     = (__bf16*)(ws + off); off += (size_t)KVROWS * CDIM * sizeof(__bf16);
    __bf16* qkv    = (__bf16*)(ws + off); off += (size_t)KVROWS * QKVN * sizeof(__bf16);
    __bf16* aout   = (__bf16*)(ws + off); off += (size_t)NTOK   * CDIM * sizeof(__bf16);
    __bf16* wqkvT  = (__bf16*)(ws + off); off += (size_t)QKVN   * CDIM * sizeof(__bf16);
    __bf16* wprojT = (__bf16*)(ws + off); off += (size_t)CDIM   * CDIM * sizeof(__bf16);

    prep_w_kernel<<<dim3(QKVN + CDIM), 256, 0, stream>>>(Wqkv, Wproj, wqkvT, wprojT);
    build_kv_kernel<<<dim3(NBLK), 256, 0, stream>>>(x, kv);
    // qkv GEMM: 67584 x 768 x 256 -> grid (528, 12)
    gemm_block_kernel<__bf16><<<dim3(KVROWS / 128, QKVN / 64), 256, 0, stream>>>(
        kv, wqkvT, bqkv, qkv, QKVN);
    attn_kernel<<<dim3(NBLK, NHEAD), 32, 0, stream>>>(qkv, amask, edge, Wgate, bgate, aout);
    // proj GEMM: 65536 x 256 x 256 -> grid (512, 4)
    gemm_block_kernel<float><<<dim3(NTOK / 128, CDIM / 64), 256, 0, stream>>>(
        aout, wprojT, bproj, out, CDIM);
}
